// Capsule_82884278878892
// MI455X (gfx1250) — matrix operand order repaired and checked by reading, not compiled, not run on hardware
//
#include <hip/hip_runtime.h>
#include <math.h>

// ---------------------------------------------------------------------------
// Capsule routing kernel for MI455X (gfx1250, wave32, WMMA)
//   s[b,(k,d)] = sum_{(n,i)} x[b,n,i] * W[n,k,d,i] * softmax_n(R)[n,k]
//   v = squash(s)  over d
// GEMM: M=256, N=512 (k*16+d), K=32768 (n*16+i).
// Precision: split-bf16 (hi+lo) for both operands; W*Rs split is precomputed
// once into workspace so the GEMM inner loop is XDL-bound, not VALU-bound.
// ---------------------------------------------------------------------------

typedef __attribute__((ext_vector_type(16))) __bf16 v16bf;
typedef __attribute__((ext_vector_type(8)))  float  v8f;

#define NCAP    2048          // n
#define KCAP    32            // k
#define DIMCAP  16            // dim (also in_dim)
#define KDIM    (NCAP*16)     // 32768  GEMM K
#define NCOL    (KCAP*DIMCAP) // 512    GEMM N
#define MROW    256           // batch  GEMM M
#define CHUNKS  16            // K split for parallelism (deterministic partials)
#define KSTEPS  64            // K-steps of 32 per chunk: 16*64*32 = 32768

__device__ __forceinline__ v8f wmma_bf16(v16bf a, v16bf b, v8f c) {
  // (neg_a, A, neg_b, B, c_mod, C, reuse_a, reuse_b)
  return __builtin_amdgcn_wmma_f32_16x16x32_bf16(false, a, false, b,
                                                 (short)0, c, false, false);
}

__device__ __forceinline__ void split_bf16(float f, __bf16& hi, __bf16& lo) {
  hi = (__bf16)f;                 // RNE to bf16
  lo = (__bf16)(f - (float)hi);   // residual, also bf16 (full fp32 exp range)
}

// Load 16 fp32 A-row elements in WMMA A-fragment order for this lane:
// elements 0..15 = row[Kbase + h*16 + 0..15]
__device__ __forceinline__ void load_a_row(const float* __restrict__ row,
                                           int Kbase, int h, float* af) {
  const float4 q0 = *(const float4*)(row + Kbase + h * 16);
  const float4 q1 = *(const float4*)(row + Kbase + h * 16 + 4);
  const float4 q2 = *(const float4*)(row + Kbase + h * 16 + 8);
  const float4 q3 = *(const float4*)(row + Kbase + h * 16 + 12);
  af[0]=q0.x;  af[1]=q0.y;  af[2]=q0.z;  af[3]=q0.w;
  af[4]=q1.x;  af[5]=q1.y;  af[6]=q1.z;  af[7]=q1.w;
  af[8]=q2.x;  af[9]=q2.y;  af[10]=q2.z; af[11]=q2.w;
  af[12]=q3.x; af[13]=q3.y; af[14]=q3.z; af[15]=q3.w;
}

// ---------------------------------------------------------------------------
// Column softmax over n for each capsule k: Rs[n,k] = softmax_n(R[:,k])
// ---------------------------------------------------------------------------
__global__ __launch_bounds__(256)
void caps_softmax_kernel(const float* __restrict__ R, float* __restrict__ Rs) {
  const int kc  = blockIdx.x;   // 0..31
  const int tid = threadIdx.x;  // 0..255
  __shared__ float red[256];

  float mx = -3.402823466e38f;
  for (int nn = tid; nn < NCAP; nn += 256)
    mx = fmaxf(mx, R[nn * KCAP + kc]);
  red[tid] = mx;
  __syncthreads();
  for (int s = 128; s > 0; s >>= 1) {
    if (tid < s) red[tid] = fmaxf(red[tid], red[tid + s]);
    __syncthreads();
  }
  mx = red[0];
  __syncthreads();

  float sum = 0.0f;
  for (int nn = tid; nn < NCAP; nn += 256)
    sum += expf(R[nn * KCAP + kc] - mx);
  red[tid] = sum;
  __syncthreads();
  for (int s = 128; s > 0; s >>= 1) {
    if (tid < s) red[tid] += red[tid + s];
    __syncthreads();
  }
  const float inv = 1.0f / red[0];

  for (int nn = tid; nn < NCAP; nn += 256)
    Rs[nn * KCAP + kc] = expf(R[nn * KCAP + kc] - mx) * inv;
}

// ---------------------------------------------------------------------------
// Preprocess: whl[(n,k,d)] = 64B block { bf16_hi(W*Rs)[i=0..15],
//                                        bf16_lo(W*Rs)[i=0..15] }
// One thread per (n,k,d): 1,048,576 threads. This is laid out so a GEMM lane's
// B-fragment pair (bh, bl) is one contiguous, fragment-ordered cache line.
// ---------------------------------------------------------------------------
__global__ __launch_bounds__(256)
void caps_prep_kernel(const float* __restrict__ W, const float* __restrict__ Rs,
                      __bf16* __restrict__ whl) {
  const int t = blockIdx.x * 256 + threadIdx.x;   // t = (n*32 + k)*16 + d
  const int k = (t >> 4) & (KCAP - 1);
  const int n = t >> 9;
  const float rs = Rs[(size_t)n * KCAP + k];

  const float* wp = W + (size_t)t * 16;
  const float4 w0 = *(const float4*)(wp + 0);
  const float4 w1 = *(const float4*)(wp + 4);
  const float4 w2 = *(const float4*)(wp + 8);
  const float4 w3 = *(const float4*)(wp + 12);
  float wf[16];
  wf[0]=w0.x;  wf[1]=w0.y;  wf[2]=w0.z;  wf[3]=w0.w;
  wf[4]=w1.x;  wf[5]=w1.y;  wf[6]=w1.z;  wf[7]=w1.w;
  wf[8]=w2.x;  wf[9]=w2.y;  wf[10]=w2.z; wf[11]=w2.w;
  wf[12]=w3.x; wf[13]=w3.y; wf[14]=w3.z; wf[15]=w3.w;

  v16bf hv, lv;
#pragma unroll
  for (int e = 0; e < 16; ++e) {
    __bf16 hi, lo;
    split_bf16(wf[e] * rs, hi, lo);
    hv[e] = hi; lv[e] = lo;
  }
  __bf16* op = whl + (size_t)t * 32;   // 64B-aligned block
  *(v16bf*)(op)      = hv;
  *(v16bf*)(op + 16) = lv;
}

// ---------------------------------------------------------------------------
// GEMM: each wave computes a 32(M) x 64(N) tile over one K-chunk.
// 1024 waves = 8 mgroups x 8 ngroups x 16 chunks. Partials to workspace.
// B fragments stream straight from whl (no VALU); only x is split in-register.
// ---------------------------------------------------------------------------
__global__ __launch_bounds__(256)
void caps_gemm_kernel(const float* __restrict__ x,
                      const __bf16* __restrict__ whl,
                      float* __restrict__ part) {
  const int lane  = threadIdx.x & 31;
  const int w     = blockIdx.x * 8 + (threadIdx.x >> 5);
  const int chunk = w & (CHUNKS - 1);
  const int ngrp  = (w >> 4) & 7;
  const int mgrp  = w >> 7;          // 0..7

  const int m  = lane & 15;          // A row-in-tile / B column (d) / C column
  const int h  = lane >> 4;          // lane half
  const int k0 = ngrp * 4;           // 4 capsule columns per wave

  const float* xrow0 = x + (size_t)(mgrp * 32 + m) * KDIM;
  const float* xrow1 = xrow0 + (size_t)16 * KDIM;

  v8f acc0[4] = {};
  v8f acc1[4] = {};

  for (int t = 0; t < KSTEPS; ++t) {
    const int T      = chunk * KSTEPS + t;
    const int Kbase  = T * 32;
    const int nprime = 2 * T + h;    // n index this lane-half covers

    // ---- A fragments (hi/lo split in registers) for both M-halves
    float af0[16], af1[16];
    load_a_row(xrow0, Kbase, h, af0);
    load_a_row(xrow1, Kbase, h, af1);
    v16bf ah0, al0, ah1, al1;
#pragma unroll
    for (int e = 0; e < 16; ++e) {
      __bf16 hi, lo;
      split_bf16(af0[e], hi, lo); ah0[e] = hi; al0[e] = lo;
      split_bf16(af1[e], hi, lo); ah1[e] = hi; al1[e] = lo;
    }

    // ---- B fragments stream directly from preprocessed whl
    const __bf16* bbase =
        whl + (((size_t)nprime * KCAP * 16) + (size_t)m) * 32;

#pragma unroll
    for (int kk = 0; kk < 4; ++kk) {
      const int kc = k0 + kk;
      const __bf16* bp = bbase + (size_t)kc * (16 * 32);
      const v16bf bh = *(const v16bf*)(bp);
      const v16bf bl = *(const v16bf*)(bp + 16);

      // split-bf16 product: x*w ~= xh*wh + xl*wh + xh*wl (xl*wl ~ 2^-32, drop)
      acc0[kk] = wmma_bf16(ah0, bh, acc0[kk]);
      acc0[kk] = wmma_bf16(al0, bh, acc0[kk]);
      acc0[kk] = wmma_bf16(ah0, bl, acc0[kk]);
      acc1[kk] = wmma_bf16(ah1, bh, acc1[kk]);
      acc1[kk] = wmma_bf16(al1, bh, acc1[kk]);
      acc1[kk] = wmma_bf16(ah1, bl, acc1[kk]);
    }
  }

  // ---- store per-chunk partials (exclusive ownership -> deterministic)
  float* pc = part + (size_t)chunk * (MROW * NCOL);
  const int row0 = mgrp * 32 + h * 8;
#pragma unroll
  for (int kk = 0; kk < 4; ++kk) {
    const int col = (k0 + kk) * 16 + m;
#pragma unroll
    for (int r = 0; r < 8; ++r) {
      pc[(size_t)(row0 + r) * NCOL + col]      = acc0[kk][r];
      pc[(size_t)(row0 + 16 + r) * NCOL + col] = acc1[kk][r];
    }
  }
}

// ---------------------------------------------------------------------------
// Reduce the 16 K-chunk partials and apply squash over the capsule dim.
// One thread per (b,k) pair: 8192 threads.
// ---------------------------------------------------------------------------
__global__ __launch_bounds__(256)
void caps_squash_kernel(const float* __restrict__ part, float* __restrict__ out) {
  const int t = blockIdx.x * 256 + threadIdx.x;  // (b,k) flat, 0..8191
  float a[16];
#pragma unroll
  for (int d = 0; d < 16; ++d) a[d] = 0.0f;

  const float* p = part + (size_t)t * 16;
  for (int c = 0; c < CHUNKS; ++c) {
    const float4* q = (const float4*)(p + (size_t)c * (MROW * NCOL));
    const float4 q0 = q[0], q1 = q[1], q2 = q[2], q3 = q[3];
    a[0]+=q0.x;  a[1]+=q0.y;  a[2]+=q0.z;  a[3]+=q0.w;
    a[4]+=q1.x;  a[5]+=q1.y;  a[6]+=q1.z;  a[7]+=q1.w;
    a[8]+=q2.x;  a[9]+=q2.y;  a[10]+=q2.z; a[11]+=q2.w;
    a[12]+=q3.x; a[13]+=q3.y; a[14]+=q3.z; a[15]+=q3.w;
  }

  float sq = 1e-7f;  // keras epsilon
#pragma unroll
  for (int d = 0; d < 16; ++d) sq += a[d] * a[d];
  const float scale = sqrtf(sq) / (1.0f + sq);

  float4* op = (float4*)(out + (size_t)t * 16);
  float4 o;
  o.x = scale*a[0];  o.y = scale*a[1];  o.z = scale*a[2];  o.w = scale*a[3];  op[0] = o;
  o.x = scale*a[4];  o.y = scale*a[5];  o.z = scale*a[6];  o.w = scale*a[7];  op[1] = o;
  o.x = scale*a[8];  o.y = scale*a[9];  o.z = scale*a[10]; o.w = scale*a[11]; op[2] = o;
  o.x = scale*a[12]; o.y = scale*a[13]; o.z = scale*a[14]; o.w = scale*a[15]; op[3] = o;
}

// ---------------------------------------------------------------------------
extern "C" void kernel_launch(void* const* d_in, const int* in_sizes, int n_in,
                              void* d_out, int out_size, void* d_ws, size_t ws_size,
                              hipStream_t stream) {
  (void)in_sizes; (void)n_in; (void)out_size; (void)ws_size;
  const float* x = (const float*)d_in[0];  // [256, 2048, 16]
  const float* W = (const float*)d_in[1];  // [2048, 32, 16, 16]
  const float* R = (const float*)d_in[2];  // [2048, 32]
  float* out = (float*)d_out;              // [256, 32, 16]

  // workspace layout:
  //   Rs   : 2048*32 f32                      =   256 KB
  //   whl  : 2048*32*16 blocks of 32 bf16     =    64 MB
  //   part : 16*256*512 f32                   =     8 MB
  float*  Rs   = (float*)d_ws;
  __bf16* whl  = (__bf16*)((char*)d_ws + (size_t)NCAP * KCAP * 4);
  float*  part = (float*)((char*)whl + (size_t)NCAP * KCAP * 16 * 32 * 2);

  caps_softmax_kernel<<<KCAP, 256, 0, stream>>>(R, Rs);
  caps_prep_kernel<<<(NCAP * KCAP * 16) / 256, 256, 0, stream>>>(W, Rs, whl);
  caps_gemm_kernel<<<128, 256, 0, stream>>>(x, whl, part);
  caps_squash_kernel<<<32, 256, 0, stream>>>(part, out);
}
